// GATModel_3_4535485465121
// MI455X (gfx1250) — matrix operand order repaired and checked by reading, not compiled, not run on hardware
//
#include <hip/hip_runtime.h>
#include <hip/hip_bf16.h>

// ---------------------------------------------------------------------------
// GATv2 (H=3 heads, C=1024) + FC readout for MI455X (gfx1250, wave32, WMMA)
// ---------------------------------------------------------------------------

typedef __attribute__((ext_vector_type(16))) __bf16 v16bf;
typedef __attribute__((ext_vector_type(8)))  __bf16 v8bf;
typedef __attribute__((ext_vector_type(8)))  float  v8f;

#define N_NODES  10000
#define E_EDGES  100000
#define ET       (E_EDGES + N_NODES)   // edges + self loops = 110000
#define F_IN     1024
#define H_HEADS  3
#define C_HID    1024
#define HC       (H_HEADS * C_HID)     // 3072
#define HC2      (2 * HC)              // 6144 (xl | xr)
#define NCLS     460
#define NC_PAD   512
#define NEG_SLOPE 0.2f

// ---------------------------------------------------------------------------
// WMMA fragment loaders (16x16x32 bf16, wave32; same contiguous k order chosen for A and B)
// A (16x32, MxK): lane half h holds K = h*16+0..15 (contiguous)   -> two 16B loads
// B (32x16, KxN): lane half h holds K = h*16+0..15 (contiguous)   -> one 32B load
// ---------------------------------------------------------------------------
__device__ __forceinline__ v16bf load_a_frag(const __bf16* row, int kbase, int half) {
    union { v16bf v; v8bf h[2]; } f;
    f.h[0] = *(const v8bf*)(row + kbase + half * 16);
    f.h[1] = *(const v8bf*)(row + kbase + half * 16 + 8);
    return f.v;
}
__device__ __forceinline__ v16bf load_b_frag(const __bf16* col, int kbase, int half) {
    return *(const v16bf*)(col + kbase + half * 16);
}

// ---------------------------------------------------------------------------
// Prep kernels: bf16 cast of x, transposed bf16 weights
// ---------------------------------------------------------------------------
__global__ void cast_x_kernel(const float* __restrict__ x, __bf16* __restrict__ xb) {
    int i = blockIdx.x * blockDim.x + threadIdx.x;
    if (i < N_NODES * F_IN) xb[i] = (__bf16)x[i];
}

// wt[col][k] = (col < HC) ? Wl[k][col] : Wr[k][col-HC]   (col-major of [Wl|Wr])
__global__ void build_wt_kernel(const float* __restrict__ Wl, const float* __restrict__ Wr,
                                __bf16* __restrict__ wt) {
    int i = blockIdx.x * blockDim.x + threadIdx.x;
    if (i >= HC2 * F_IN) return;
    int col = i / F_IN;
    int k   = i - col * F_IN;
    float v = (col < HC) ? Wl[(size_t)k * HC + col] : Wr[(size_t)k * HC + (col - HC)];
    wt[i] = (__bf16)v;
}

// wfct[n][k] = (n < NCLS) ? Wfc[k][n] : 0   (zero-padded to NC_PAD rows)
__global__ void build_wfct_kernel(const float* __restrict__ Wfc, __bf16* __restrict__ wfct) {
    int i = blockIdx.x * blockDim.x + threadIdx.x;
    if (i >= NC_PAD * F_IN) return;
    int n = i / F_IN;
    int k = i - n * F_IN;
    wfct[i] = (__bf16)((n < NCLS) ? Wfc[(size_t)k * NCLS + n] : 0.0f);
}

// ---------------------------------------------------------------------------
// GEMM 1: xlr[N,6144](bf16) = xb[N,1024] @ [Wl|Wr] + [bl|br]
// Block = 8 waves; each wave computes a 32(M) x 64(N) tile -> 8 WMMAs / K-step
// ---------------------------------------------------------------------------
__global__ void __launch_bounds__(256) gemm_xlr_kernel(
    const __bf16* __restrict__ xb, const __bf16* __restrict__ wt,
    const float* __restrict__ bl, const float* __restrict__ br,
    __bf16* __restrict__ xlrb) {
    const int lane = threadIdx.x & 31;
    const int wave = threadIdx.x >> 5;
    const int half = lane >> 4;
    const int l16  = lane & 15;
    const int colBase = blockIdx.x * 512 + wave * 64;  // 8 waves * 64 cols
    const int rowBase = blockIdx.y * 32;

    int ra0 = rowBase + l16;      if (ra0 >= N_NODES) ra0 = N_NODES - 1;
    int ra1 = rowBase + 16 + l16; if (ra1 >= N_NODES) ra1 = N_NODES - 1;
    const __bf16* a0 = xb + (size_t)ra0 * F_IN;
    const __bf16* a1 = xb + (size_t)ra1 * F_IN;
    const __bf16* bp[4];
#pragma unroll
    for (int j = 0; j < 4; ++j) bp[j] = wt + (size_t)(colBase + j * 16 + l16) * F_IN;

    const v8f vzero = {};
    v8f acc[2][4];
#pragma unroll
    for (int i = 0; i < 2; ++i)
#pragma unroll
        for (int j = 0; j < 4; ++j) acc[i][j] = vzero;

    for (int k = 0; k < F_IN; k += 32) {
        __builtin_prefetch(a0 + k + 128, 0, 1);
        __builtin_prefetch(a1 + k + 128, 0, 1);
        v16bf A0 = load_a_frag(a0, k, half);
        v16bf A1 = load_a_frag(a1, k, half);
#pragma unroll
        for (int j = 0; j < 4; ++j) {
            v16bf B = load_b_frag(bp[j], k, half);
            acc[0][j] = __builtin_amdgcn_wmma_f32_16x16x32_bf16(
                false, A0, false, B, (short)0, acc[0][j], false, false);
            acc[1][j] = __builtin_amdgcn_wmma_f32_16x16x32_bf16(
                false, A1, false, B, (short)0, acc[1][j], false, false);
        }
    }

#pragma unroll
    for (int i = 0; i < 2; ++i) {
#pragma unroll
        for (int r = 0; r < 8; ++r) {
            int row = rowBase + i * 16 + half * 8 + r;
            if (row < N_NODES) {
#pragma unroll
                for (int j = 0; j < 4; ++j) {
                    int col = colBase + j * 16 + l16;
                    float b = (col < HC) ? bl[col] : br[col - HC];
                    xlrb[(size_t)row * HC2 + col] = (__bf16)(acc[i][j][r] + b);
                }
            }
        }
    }
}

// ---------------------------------------------------------------------------
// GEMM 2: out[N,460](f32) = hmb[N,1024] @ Wfc + bfc   (NC padded to 512)
// ---------------------------------------------------------------------------
__global__ void __launch_bounds__(256) gemm_fc_kernel(
    const __bf16* __restrict__ hmb, const __bf16* __restrict__ wfct,
    const float* __restrict__ bfc, float* __restrict__ out) {
    const int lane = threadIdx.x & 31;
    const int wave = threadIdx.x >> 5;
    const int half = lane >> 4;
    const int l16  = lane & 15;
    const int colBase = blockIdx.x * 512 + wave * 64;  // grid.x = 1 (512 cols)
    const int rowBase = blockIdx.y * 32;

    int ra0 = rowBase + l16;      if (ra0 >= N_NODES) ra0 = N_NODES - 1;
    int ra1 = rowBase + 16 + l16; if (ra1 >= N_NODES) ra1 = N_NODES - 1;
    const __bf16* a0 = hmb + (size_t)ra0 * C_HID;
    const __bf16* a1 = hmb + (size_t)ra1 * C_HID;
    const __bf16* bp[4];
#pragma unroll
    for (int j = 0; j < 4; ++j) bp[j] = wfct + (size_t)(colBase + j * 16 + l16) * C_HID;

    const v8f vzero = {};
    v8f acc[2][4];
#pragma unroll
    for (int i = 0; i < 2; ++i)
#pragma unroll
        for (int j = 0; j < 4; ++j) acc[i][j] = vzero;

    for (int k = 0; k < C_HID; k += 32) {
        v16bf A0 = load_a_frag(a0, k, half);
        v16bf A1 = load_a_frag(a1, k, half);
#pragma unroll
        for (int j = 0; j < 4; ++j) {
            v16bf B = load_b_frag(bp[j], k, half);
            acc[0][j] = __builtin_amdgcn_wmma_f32_16x16x32_bf16(
                false, A0, false, B, (short)0, acc[0][j], false, false);
            acc[1][j] = __builtin_amdgcn_wmma_f32_16x16x32_bf16(
                false, A1, false, B, (short)0, acc[1][j], false, false);
        }
    }

#pragma unroll
    for (int i = 0; i < 2; ++i) {
#pragma unroll
        for (int r = 0; r < 8; ++r) {
            int row = rowBase + i * 16 + half * 8 + r;
            if (row < N_NODES) {
#pragma unroll
                for (int j = 0; j < 4; ++j) {
                    int col = colBase + j * 16 + l16;
                    if (col < NCLS)
                        out[(size_t)row * NCLS + col] = acc[i][j][r] + bfc[col];
                }
            }
        }
    }
}

// ---------------------------------------------------------------------------
// Edge phase
// ---------------------------------------------------------------------------
__device__ __forceinline__ unsigned f2ord(float f) {
    unsigned u = __float_as_uint(f);
    return (u & 0x80000000u) ? ~u : (u | 0x80000000u);
}
__device__ __forceinline__ float ord2f(unsigned v) {
    unsigned u = (v & 0x80000000u) ? (v & 0x7FFFFFFFu) : ~v;
    return __uint_as_float(u);
}

// score[e,h] = att[h] . leaky_relu(xl[src,h] + xr[dst,h]); one wave per edge
__global__ void __launch_bounds__(256) edge_score_kernel(
    const __bf16* __restrict__ xlrb, const long long* __restrict__ ei,
    const float* __restrict__ att, float* __restrict__ score) {
    int e    = blockIdx.x * 8 + (threadIdx.x >> 5);
    int lane = threadIdx.x & 31;
    if (e >= ET) return;
    int src, dst;
    if (e < E_EDGES) { src = (int)ei[2 * e]; dst = (int)ei[2 * e + 1]; }
    else             { src = dst = e - E_EDGES; }
    const __bf16* xl = xlrb + (size_t)src * HC2;
    const __bf16* xr = xlrb + (size_t)dst * HC2 + HC;
    for (int h = 0; h < H_HEADS; ++h) {
        float acc = 0.0f;
        for (int c = lane; c < C_HID; c += 32) {
            float s = (float)xl[h * C_HID + c] + (float)xr[h * C_HID + c];
            s = (s > 0.0f) ? s : NEG_SLOPE * s;
            acc += s * att[h * C_HID + c];
        }
#pragma unroll
        for (int off = 16; off > 0; off >>= 1) acc += __shfl_xor(acc, off, 32);
        if (lane == 0) score[(size_t)e * H_HEADS + h] = acc;
    }
}

__global__ void init_kernel(unsigned* __restrict__ smax_u, float* __restrict__ denom,
                            float* __restrict__ hagg) {
    int i = blockIdx.x * blockDim.x + threadIdx.x;
    if (i < N_NODES * C_HID) hagg[i] = 0.0f;
    if (i < N_NODES * H_HEADS) { smax_u[i] = 0u; denom[i] = 0.0f; }
}

__global__ void smax_kernel(const float* __restrict__ score, const long long* __restrict__ ei,
                            unsigned* __restrict__ smax_u) {
    int idx = blockIdx.x * blockDim.x + threadIdx.x;
    if (idx >= ET * H_HEADS) return;
    int e = idx / H_HEADS;
    int h = idx - e * H_HEADS;
    int dst = (e < E_EDGES) ? (int)ei[2 * e + 1] : e - E_EDGES;
    atomicMax(&smax_u[dst * H_HEADS + h], f2ord(score[idx]));
}

__global__ void es_kernel(const float* __restrict__ score, const long long* __restrict__ ei,
                          const unsigned* __restrict__ smax_u,
                          float* __restrict__ es, float* __restrict__ denom) {
    int idx = blockIdx.x * blockDim.x + threadIdx.x;
    if (idx >= ET * H_HEADS) return;
    int e = idx / H_HEADS;
    int h = idx - e * H_HEADS;
    int dst = (e < E_EDGES) ? (int)ei[2 * e + 1] : e - E_EDGES;
    float m = ord2f(smax_u[dst * H_HEADS + h]);
    float v = __expf(score[idx] - m);
    es[idx] = v;
    atomicAdd(&denom[dst * H_HEADS + h], v);
}

// hagg[dst,c] += sum_h alpha[e,h] * xl[src,h,c]   (head-sum; /H folded later)
__global__ void __launch_bounds__(256) aggregate_kernel(
    const __bf16* __restrict__ xlrb, const long long* __restrict__ ei,
    const float* __restrict__ es, const float* __restrict__ denom,
    float* __restrict__ hagg) {
    int e = blockIdx.x;
    int src, dst;
    if (e < E_EDGES) { src = (int)ei[2 * e]; dst = (int)ei[2 * e + 1]; }
    else             { src = dst = e - E_EDGES; }
    float a0 = es[e * 3 + 0] / (denom[dst * 3 + 0] + 1e-16f);
    float a1 = es[e * 3 + 1] / (denom[dst * 3 + 1] + 1e-16f);
    float a2 = es[e * 3 + 2] / (denom[dst * 3 + 2] + 1e-16f);
    const __bf16* xl = xlrb + (size_t)src * HC2;
    for (int c = threadIdx.x; c < C_HID; c += 256) {
        float v = a0 * (float)xl[c]
                + a1 * (float)xl[C_HID + c]
                + a2 * (float)xl[2 * C_HID + c];
        atomicAdd(&hagg[(size_t)dst * C_HID + c], v);
    }
}

// hmb = bf16( hagg/H + bias )
__global__ void finalize_kernel(const float* __restrict__ hagg, const float* __restrict__ bias,
                                __bf16* __restrict__ hmb) {
    int i = blockIdx.x * blockDim.x + threadIdx.x;
    if (i >= N_NODES * C_HID) return;
    hmb[i] = (__bf16)(hagg[i] * (1.0f / 3.0f) + bias[i & (C_HID - 1)]);
}

__global__ void copy_kernel(const float* __restrict__ src, float* __restrict__ dst, int n) {
    int i = blockIdx.x * blockDim.x + threadIdx.x;
    if (i < n) dst[i] = src[i];
}

// ---------------------------------------------------------------------------
// Launcher
// ---------------------------------------------------------------------------
extern "C" void kernel_launch(void* const* d_in, const int* in_sizes, int n_in,
                              void* d_out, int out_size, void* d_ws, size_t ws_size,
                              hipStream_t stream) {
    (void)in_sizes; (void)n_in; (void)out_size; (void)ws_size;
    const float*     x    = (const float*)d_in[0];
    const long long* ei   = (const long long*)d_in[1];
    const float*     Wl   = (const float*)d_in[2];
    const float*     bl   = (const float*)d_in[3];
    const float*     Wr   = (const float*)d_in[4];
    const float*     br   = (const float*)d_in[5];
    const float*     att  = (const float*)d_in[6];
    const float*     bias = (const float*)d_in[7];
    const float*     Wfc  = (const float*)d_in[8];
    const float*     bfc  = (const float*)d_in[9];
    const float*     exps  = (const float*)d_in[10];
    const float*     expsc = (const float*)d_in[11];
    float* out = (float*)d_out;

    char* ws = (char*)d_ws;
    size_t off = 0;
    auto wsalloc = [&](size_t bytes) -> void* {
        void* p = ws + off;
        off += (bytes + 255) & ~(size_t)255;
        return p;
    };
    __bf16*   xb     = (__bf16*)  wsalloc((size_t)N_NODES * F_IN * 2);
    __bf16*   wt     = (__bf16*)  wsalloc((size_t)HC2 * F_IN * 2);
    __bf16*   xlrb   = (__bf16*)  wsalloc((size_t)N_NODES * HC2 * 2);
    __bf16*   hmb    = (__bf16*)  wsalloc((size_t)N_NODES * C_HID * 2);
    __bf16*   wfct   = (__bf16*)  wsalloc((size_t)NC_PAD * F_IN * 2);
    float*    score  = (float*)   wsalloc((size_t)ET * H_HEADS * 4);
    float*    es     = (float*)   wsalloc((size_t)ET * H_HEADS * 4);
    unsigned* smax_u = (unsigned*)wsalloc((size_t)N_NODES * H_HEADS * 4);
    float*    denom  = (float*)   wsalloc((size_t)N_NODES * H_HEADS * 4);
    float*    hagg   = (float*)   wsalloc((size_t)N_NODES * C_HID * 4);

    // 1) precision / layout prep
    cast_x_kernel<<<(N_NODES * F_IN + 255) / 256, 256, 0, stream>>>(x, xb);
    build_wt_kernel<<<(HC2 * F_IN + 255) / 256, 256, 0, stream>>>(Wl, Wr, wt);
    build_wfct_kernel<<<(NC_PAD * F_IN + 255) / 256, 256, 0, stream>>>(Wfc, wfct);

    // 2) projection GEMM (WMMA): xlr = x @ [Wl|Wr] + [bl|br]
    gemm_xlr_kernel<<<dim3(HC2 / 512, (N_NODES + 31) / 32), 256, 0, stream>>>(
        xb, wt, bl, br, xlrb);

    // 3) edge scores
    edge_score_kernel<<<(ET + 7) / 8, 256, 0, stream>>>(xlrb, ei, att, score);

    // 4) segment softmax + aggregation
    init_kernel<<<(N_NODES * C_HID + 255) / 256, 256, 0, stream>>>(smax_u, denom, hagg);
    smax_kernel<<<(ET * H_HEADS + 255) / 256, 256, 0, stream>>>(score, ei, smax_u);
    es_kernel<<<(ET * H_HEADS + 255) / 256, 256, 0, stream>>>(score, ei, smax_u, es, denom);
    aggregate_kernel<<<ET, 256, 0, stream>>>(xlrb, ei, es, denom, hagg);
    finalize_kernel<<<(N_NODES * C_HID + 255) / 256, 256, 0, stream>>>(hagg, bias, hmb);

    // 5) FC readout GEMM (WMMA) -> d_out[0 : N*NC]
    gemm_fc_kernel<<<dim3(NC_PAD / 512, (N_NODES + 31) / 32), 256, 0, stream>>>(
        hmb, wfct, bfc, out);

    // 6) pass-through outputs
    copy_kernel<<<(N_NODES * NCLS + 255) / 256, 256, 0, stream>>>(
        exps, out + (size_t)N_NODES * NCLS, N_NODES * NCLS);
    copy_kernel<<<(128 * NCLS + 255) / 256, 256, 0, stream>>>(
        expsc, out + 2 * (size_t)N_NODES * NCLS, 128 * NCLS);
}
